// MultiHeadAttention_24790551233207
// MI455X (gfx1250) — matrix operand order repaired and checked by reading, not compiled, not run on hardware
//
#include <hip/hip_runtime.h>
#include <math.h>
#include <stdint.h>

// ---------------------------------------------------------------------------
// MHA forward for MI455X (gfx1250): bf16 WMMA everywhere, fp32 accumulate.
// ---------------------------------------------------------------------------

typedef __attribute__((ext_vector_type(16))) __bf16          v16bf;
typedef __attribute__((ext_vector_type(8)))  float           v8f;
typedef __attribute__((ext_vector_type(8)))  unsigned short  u16x8;
typedef __attribute__((ext_vector_type(16))) unsigned short  u16x16;

#define D_MODEL 1024
#define N_HEAD  16
#define DH      64
#define BATCH   2
#define SEQ     2048
#define ROWS    (BATCH * SEQ)   // 4096

// fp32 -> bf16, round-to-nearest-even
__device__ __forceinline__ unsigned short f2bf(float f) {
    unsigned int u = __float_as_uint(f);
    unsigned int r = u + 0x7fffu + ((u >> 16) & 1u);
    return (unsigned short)(r >> 16);
}

__device__ __forceinline__ v8f wmma_bf16(v16bf a, v16bf b, v8f c) {
    // (neg_a, A, neg_b, B, c_mod, C, reuse_a, reuse_b)
    return __builtin_amdgcn_wmma_f32_16x16x32_bf16(false, a, false, b,
                                                   (short)0, c, false, false);
}

// A fragment (16x32 bf16, M x K). Memory view: s[row*ld + k], row-major in K.
// lane l<16: M=l,    K in {0..7} then {16..23}
// lane l>=16: M=l-16, K in {8..15} then {24..31}
__device__ __forceinline__ v16bf ld_frag_a(const unsigned short* s, int ld) {
    int l = threadIdx.x & 31;
    const unsigned short* p = s + (l & 15) * ld + ((l < 16) ? 0 : 8);
    u16x8 lo = *(const u16x8*)(p);
    u16x8 hi = *(const u16x8*)(p + 16);
    u16x16 v = __builtin_shufflevector(lo, hi, 0,1,2,3,4,5,6,7,8,9,10,11,12,13,14,15);
    return __builtin_bit_cast(v16bf, v);
}

// B fragment (32x16 bf16, K x N). Memory view: s[n*ld + k]  (column n, reduction k
// contiguous). lane l<16: N=l, K in {0..7} then {16..23} ; lane l>=16: N=l-16, K in {8..15} then {24..31}
__device__ __forceinline__ v16bf ld_frag_b(const unsigned short* s, int ld) {
    int l = threadIdx.x & 31;
    const unsigned short* p = s + (l & 15) * ld + ((l < 16) ? 0 : 8);
    u16x8 lo = *(const u16x8*)(p);
    u16x8 hi = *(const u16x8*)(p + 16);
    u16x16 v = __builtin_shufflevector(lo, hi, 0,1,2,3,4,5,6,7,8,9,10,11,12,13,14,15);
    return __builtin_bit_cast(v16bf, v);
}

// ---------------------------------------------------------------------------
// Kernel 1: fused QKV projection.  out[m,n] = sum_k X[m,k] * W[n,k]  (x @ W.T)
// z = blockIdx.z selects (X, W, dest/layout): 0->Q, 1->K, 2->V(transposed)
// Block tile 128x64, 8 waves of 32x32, K step 32, LDS-staged bf16.
// ---------------------------------------------------------------------------
__device__ __forceinline__ void st_qkv(int z, unsigned short* Qb, unsigned short* Kb,
                                       unsigned short* Vt, int m, int n, float v) {
    int b = m >> 11, t = m & 2047, h = n >> 6, d = n & 63;
    unsigned short x = f2bf(v);
    if (z == 2)      Vt[((size_t)((b * N_HEAD + h) * DH + d)) * SEQ + t]      = x;
    else if (z == 1) Kb[(((size_t)(b * N_HEAD + h) * SEQ + t)) * DH + d]      = x;
    else             Qb[(((size_t)(b * N_HEAD + h) * SEQ + t)) * DH + d]      = x;
}

__global__ __launch_bounds__(256) void qkv_proj(const float* __restrict__ Xq,
                                                const float* __restrict__ Xkv,
                                                const float* __restrict__ Wq,
                                                const float* __restrict__ Wk,
                                                const float* __restrict__ Wv,
                                                unsigned short* __restrict__ Qb,
                                                unsigned short* __restrict__ Kb,
                                                unsigned short* __restrict__ Vt) {
    __shared__ __attribute__((aligned(16))) unsigned short As[128 * 32];
    __shared__ __attribute__((aligned(16))) unsigned short Bs[64 * 32];

    const int tid = threadIdx.x, w = tid >> 5, l = tid & 31;
    const int z = blockIdx.z;
    const float* X = (z == 0) ? Xq : Xkv;
    const float* W = (z == 0) ? Wq : ((z == 1) ? Wk : Wv);
    const int m0 = blockIdx.y * 128;
    const int n0 = blockIdx.x * 64;
    const int wr = (w >> 1) * 32, wc = (w & 1) * 32;

    v8f c00 = {}, c01 = {}, c10 = {}, c11 = {};

    for (int kt = 0; kt < 32; ++kt) {
        const int k0 = kt * 32;
        // stage A tile (128x32) as bf16: 1024 float4 chunks, 4 per thread
#pragma unroll
        for (int i = 0; i < 4; ++i) {
            int c = tid + i * 256;
            int row = c >> 3, kc = (c & 7) * 4;
            float4 v = *(const float4*)(X + (size_t)(m0 + row) * D_MODEL + k0 + kc);
            unsigned short* d = &As[row * 32 + kc];
            d[0] = f2bf(v.x); d[1] = f2bf(v.y); d[2] = f2bf(v.z); d[3] = f2bf(v.w);
        }
        // stage B tile (64x32) as bf16: 512 float4 chunks, 2 per thread
#pragma unroll
        for (int i = 0; i < 2; ++i) {
            int c = tid + i * 256;
            int row = c >> 3, kc = (c & 7) * 4;
            float4 v = *(const float4*)(W + (size_t)(n0 + row) * D_MODEL + k0 + kc);
            unsigned short* d = &Bs[row * 32 + kc];
            d[0] = f2bf(v.x); d[1] = f2bf(v.y); d[2] = f2bf(v.z); d[3] = f2bf(v.w);
        }
        if (kt + 1 < 32)  // hint next A tile into cache (global_prefetch_b8)
            __builtin_prefetch((const void*)(X + (size_t)(m0 + (tid >> 3)) * D_MODEL + k0 + 32), 0, 1);
        __syncthreads();

        v16bf a0 = ld_frag_a(As + (size_t)wr * 32, 32);
        v16bf a1 = ld_frag_a(As + (size_t)(wr + 16) * 32, 32);
        v16bf b0 = ld_frag_b(Bs + (size_t)wc * 32, 32);
        v16bf b1 = ld_frag_b(Bs + (size_t)(wc + 16) * 32, 32);
        c00 = wmma_bf16(a0, b0, c00);
        c01 = wmma_bf16(a0, b1, c01);
        c10 = wmma_bf16(a1, b0, c10);
        c11 = wmma_bf16(a1, b1, c11);
        __syncthreads();
    }

    const int roff = (l < 16) ? 0 : 8, cc = l & 15;
#pragma unroll
    for (int g = 0; g < 8; ++g) {
        int mA = m0 + wr + g + roff, mB = mA + 16;
        int nA = n0 + wc + cc,       nB = nA + 16;
        st_qkv(z, Qb, Kb, Vt, mA, nA, c00[g]);
        st_qkv(z, Qb, Kb, Vt, mA, nB, c01[g]);
        st_qkv(z, Qb, Kb, Vt, mB, nA, c10[g]);
        st_qkv(z, Qb, Kb, Vt, mB, nB, c11[g]);
    }
}

// ---------------------------------------------------------------------------
// Kernel 2: flash-style causal attention. One wave per 16-query tile; 8 waves
// per block share one (b,h); uniform trip count so __syncthreads is legal.
// ---------------------------------------------------------------------------
__global__ __launch_bounds__(256) void attn(const unsigned short* __restrict__ Qb,
                                            const unsigned short* __restrict__ Kb,
                                            const unsigned short* __restrict__ Vt,
                                            const int* __restrict__ kvmask,
                                            unsigned short* __restrict__ yb) {
    __shared__ __attribute__((aligned(16))) float          Ssc[8][16 * 32];
    __shared__ __attribute__((aligned(16))) unsigned short Pt[8][16 * 32];
    __shared__ float rowa[8][16];
    __shared__ int   smask[8][32];

    const int tid = threadIdx.x, w = tid >> 5, l = tid & 31;
    const int wid = blockIdx.x * 8 + w;
    const int qt = wid & 127;            // query tile (0..127)
    const int bh = wid >> 7;             // 0..31
    const int b = bh >> 4, h = bh & 15;
    const int q0 = qt * 16;
    const int qtbase = (blockIdx.x * 8) & 127;
    const int nsteps = ((qtbase + 8) * 16 + 31) >> 5;   // uniform across block

    const unsigned short* Qp = Qb + ((size_t)bh * SEQ + q0) * DH;
    const v16bf qa0 = ld_frag_a(Qp, DH);        // d = 0..31
    const v16bf qa1 = ld_frag_a(Qp + 32, DH);   // d = 32..63

    v8f y0 = {}, y1 = {}, y2 = {}, y3 = {};
    float mrow = -__builtin_inff(), lrow = 0.f;   // valid on lanes 0..15 (row = l)

    float*          Sp = Ssc[w];
    unsigned short* Pp = Pt[w];
    const int* mp = kvmask + b * SEQ;
    const int roff = (l < 16) ? 0 : 8, cc = l & 15;

    for (int st = 0; st < nsteps; ++st) {
        const int k0 = st * 32;
        smask[w][l] = mp[k0 + l];

        // S = Q . K^T   (two 16x16 tiles, K-dim 64 split in 2)
        const unsigned short* Kp = Kb + ((size_t)bh * SEQ + k0) * DH;
        v8f z8 = {};
        v8f s0 = wmma_bf16(qa0, ld_frag_b(Kp, DH), z8);
        s0     = wmma_bf16(qa1, ld_frag_b(Kp + 32, DH), s0);
        v8f s1 = wmma_bf16(qa0, ld_frag_b(Kp + 16 * DH, DH), z8);
        s1     = wmma_bf16(qa1, ld_frag_b(Kp + 16 * DH + 32, DH), s1);

#pragma unroll
        for (int g = 0; g < 8; ++g) {
            Sp[(g + roff) * 32 + cc]      = s0[g];
            Sp[(g + roff) * 32 + cc + 16] = s1[g];
        }
        __syncthreads();

        // online softmax: lane r (<16) owns query row q0+r
        if (l < 16) {
            const int r = l;
            const float mo = mrow;
            float vmax = -__builtin_inff();
#pragma unroll
            for (int j = 0; j < 32; ++j) {
                int kk = k0 + j;
                bool valid = (kk <= q0 + r) && (smask[w][j] != 0);
                float x = valid ? Sp[r * 32 + j] * 0.125f : -__builtin_inff();
                vmax = fmaxf(vmax, x);
            }
            float mn = fmaxf(mo, vmax);
            float alpha = (mn == -__builtin_inff()) ? 1.f : __expf(mo - mn);
            float psum = 0.f;
#pragma unroll
            for (int j = 0; j < 32; ++j) {
                int kk = k0 + j;
                bool valid = (kk <= q0 + r) && (smask[w][j] != 0);
                float x = valid ? Sp[r * 32 + j] * 0.125f : -__builtin_inff();
                float p = (mn == -__builtin_inff()) ? 0.f : __expf(x - mn);
                Pp[r * 32 + j] = f2bf(p);
                psum += p;
            }
            lrow = lrow * alpha + psum;
            mrow = mn;
            rowa[w][r] = alpha;
        }
        __syncthreads();

        // rescale accumulators, then y += P . V  (4 tiles over dh)
#pragma unroll
        for (int g = 0; g < 8; ++g) {
            float av = rowa[w][g + roff];
            y0[g] *= av; y1[g] *= av; y2[g] *= av; y3[g] *= av;
        }
        v16bf pa = ld_frag_a(Pp, 32);
        const unsigned short* Vp = Vt + (size_t)bh * DH * SEQ + k0;  // [d][t]
        y0 = wmma_bf16(pa, ld_frag_b(Vp + (size_t)0  * SEQ, SEQ), y0);
        y1 = wmma_bf16(pa, ld_frag_b(Vp + (size_t)16 * SEQ, SEQ), y1);
        y2 = wmma_bf16(pa, ld_frag_b(Vp + (size_t)32 * SEQ, SEQ), y2);
        y3 = wmma_bf16(pa, ld_frag_b(Vp + (size_t)48 * SEQ, SEQ), y3);
        __syncthreads();
    }

    // finalize: divide by softmax denominator, store with the reference's
    // faithful reshape:  r = h*128 + 2*d + (t>>10), c = t & 1023
    if (l < 16) rowa[w][l] = lrow;
    __syncthreads();
#pragma unroll
    for (int g = 0; g < 8; ++g) {
        float inv = 1.f / rowa[w][g + roff];
        int t = q0 + g + roff;
        int rbase = h * 128 + (t >> 10);
        size_t obase = ((size_t)b * SEQ) * D_MODEL + (t & 1023);
        yb[obase + (size_t)(rbase + 2 * (0  + cc)) * D_MODEL] = f2bf(y0[g] * inv);
        yb[obase + (size_t)(rbase + 2 * (16 + cc)) * D_MODEL] = f2bf(y1[g] * inv);
        yb[obase + (size_t)(rbase + 2 * (32 + cc)) * D_MODEL] = f2bf(y2[g] * inv);
        yb[obase + (size_t)(rbase + 2 * (48 + cc)) * D_MODEL] = f2bf(y3[g] * inv);
    }
}

// ---------------------------------------------------------------------------
// Kernel 3: output projection  out = yb @ Wo^T  (bf16 A, fp32 result)
// A tile staged with the gfx1250 async Global->LDS pipe (ASYNCcnt).
// ---------------------------------------------------------------------------
__global__ __launch_bounds__(256) void out_proj(const unsigned short* __restrict__ yb,
                                                const float* __restrict__ Wo,
                                                float* __restrict__ out) {
    __shared__ __attribute__((aligned(16))) unsigned short As[128 * 32];
    __shared__ __attribute__((aligned(16))) unsigned short Bs[64 * 32];

    const int tid = threadIdx.x, w = tid >> 5, l = tid & 31;
    const int m0 = blockIdx.y * 128;
    const int n0 = blockIdx.x * 64;
    const int wr = (w >> 1) * 32, wc = (w & 1) * 32;

    v8f c00 = {}, c01 = {}, c10 = {}, c11 = {};

    for (int kt = 0; kt < 32; ++kt) {
        const int k0 = kt * 32;
        // A already bf16: async-copy 512 x 16B chunks straight into LDS
        // (generic LDS pointer: low 32 bits carry the logical LDS offset)
#pragma unroll
        for (int i = 0; i < 2; ++i) {
            int c = tid + i * 256;
            int row = c >> 2, kc = (c & 3) * 8;
            unsigned lds_addr = (unsigned)(uintptr_t)(&As[row * 32 + kc]);
            const unsigned short* g = yb + (size_t)(m0 + row) * D_MODEL + k0 + kc;
            asm volatile("global_load_async_to_lds_b128 %0, %1, off"
                         :: "v"(lds_addr), "v"(g) : "memory");
        }
#pragma unroll
        for (int i = 0; i < 2; ++i) {
            int c = tid + i * 256;
            int row = c >> 3, kc = (c & 7) * 4;
            float4 v = *(const float4*)(Wo + (size_t)(n0 + row) * D_MODEL + k0 + kc);
            unsigned short* d = &Bs[row * 32 + kc];
            d[0] = f2bf(v.x); d[1] = f2bf(v.y); d[2] = f2bf(v.z); d[3] = f2bf(v.w);
        }
        asm volatile("s_wait_asynccnt 0x0" ::: "memory");
        __syncthreads();

        v16bf a0 = ld_frag_a(As + (size_t)wr * 32, 32);
        v16bf a1 = ld_frag_a(As + (size_t)(wr + 16) * 32, 32);
        v16bf b0 = ld_frag_b(Bs + (size_t)wc * 32, 32);
        v16bf b1 = ld_frag_b(Bs + (size_t)(wc + 16) * 32, 32);
        c00 = wmma_bf16(a0, b0, c00);
        c01 = wmma_bf16(a0, b1, c01);
        c10 = wmma_bf16(a1, b0, c10);
        c11 = wmma_bf16(a1, b1, c11);
        __syncthreads();
    }

    const int roff = (l < 16) ? 0 : 8, cc = l & 15;
#pragma unroll
    for (int g = 0; g < 8; ++g) {
        int mA = m0 + wr + g + roff, mB = mA + 16;
        out[(size_t)mA * D_MODEL + n0 + wc + cc]      = c00[g];
        out[(size_t)mA * D_MODEL + n0 + wc + 16 + cc] = c01[g];
        out[(size_t)mB * D_MODEL + n0 + wc + cc]      = c10[g];
        out[(size_t)mB * D_MODEL + n0 + wc + 16 + cc] = c11[g];
    }
}

// ---------------------------------------------------------------------------
extern "C" void kernel_launch(void* const* d_in, const int* in_sizes, int n_in,
                              void* d_out, int out_size, void* d_ws, size_t ws_size,
                              hipStream_t stream) {
    (void)in_sizes; (void)n_in; (void)out_size; (void)ws_size;
    const float* q_raw  = (const float*)d_in[0];
    const float* kv_raw = (const float*)d_in[1];
    const int*   kvmask = (const int*)d_in[2];
    const float* Wq     = (const float*)d_in[3];
    const float* Wk     = (const float*)d_in[4];
    const float* Wv     = (const float*)d_in[5];
    const float* Wo     = (const float*)d_in[6];
    float*       out    = (float*)d_out;

    const size_t NELEM = (size_t)BATCH * SEQ * D_MODEL;    // 4M elements
    unsigned short* Qb = (unsigned short*)d_ws;
    unsigned short* Kb = Qb + NELEM;
    unsigned short* Vt = Kb + NELEM;
    unsigned short* yb = Vt + NELEM;

    qkv_proj<<<dim3(D_MODEL / 64, ROWS / 128, 3), 256, 0, stream>>>(
        q_raw, kv_raw, Wq, Wk, Wv, Qb, Kb, Vt);
    attn<<<dim3((BATCH * N_HEAD * (SEQ / 16)) / 8), 256, 0, stream>>>(
        Qb, Kb, Vt, kvmask, yb);
    out_proj<<<dim3(D_MODEL / 64, ROWS / 128), 256, 0, stream>>>(yb, Wo, out);
}
